// MGNN_46789373723079
// MI455X (gfx1250) — matrix operand order repaired and checked by reading, not compiled, not run on hardware
//
#include <hip/hip_runtime.h>

typedef __bf16 bf16;
typedef __attribute__((ext_vector_type(16))) __bf16 v16bf;
typedef __attribute__((ext_vector_type(8)))  __bf16 bf16x8;
typedef __attribute__((ext_vector_type(4)))  __bf16 bf16x4;
typedef __attribute__((ext_vector_type(8)))  float  v8f;
typedef __attribute__((ext_vector_type(4)))  float  f32x4;
typedef __attribute__((ext_vector_type(4)))  int    i32x4;

#define HID 256

// Total-order float<->int encoding: int compare on encoded == float compare.
__device__ __forceinline__ int   enc_f32(float f) { int i = __float_as_int(f); return i ^ ((i >> 31) & 0x7FFFFFFF); }
__device__ __forceinline__ float dec_f32(int s)   { int i = s ^ ((s >> 31) & 0x7FFFFFFF); return __int_as_float(i); }
#define ENC_NEG_INF 0x807FFFFF   // enc_f32(-inf)

// ---------------- elementwise helpers ----------------

__global__ __launch_bounds__(256) void fill_i32(int* __restrict__ p, int v, long long n) {
    long long gid = (long long)blockIdx.x * blockDim.x + threadIdx.x;
    if (gid < n) p[gid] = v;
}

// f32 -> bf16 (4-wide)
__global__ __launch_bounds__(256) void cvt_bf16x4(const float* __restrict__ in, bf16* __restrict__ out,
                                                  long long n4) {
    long long gid = (long long)blockIdx.x * blockDim.x + threadIdx.x;
    if (gid >= n4) return;
    f32x4 v = ((const f32x4*)in)[gid];
    bf16x4 o;
    #pragma unroll
    for (int i = 0; i < 4; i++) o[i] = (bf16)v[i];
    ((bf16x4*)out)[gid] = o;
}

// encoded-int agg -> bf16, mapping -inf (empty segment) -> 0 like PyG
__global__ __launch_bounds__(256) void cvt_agg_bf16x4(const int* __restrict__ in, bf16* __restrict__ out,
                                                      long long n4) {
    long long gid = (long long)blockIdx.x * blockDim.x + threadIdx.x;
    if (gid >= n4) return;
    i32x4 s = ((const i32x4*)in)[gid];
    bf16x4 o;
    #pragma unroll
    for (int i = 0; i < 4; i++) {
        float f = dec_f32(s[i]);
        if (f == -__builtin_inff()) f = 0.0f;
        o[i] = (bf16)f;
    }
    ((bf16x4*)out)[gid] = o;
}

// ---------------- scatter max (segment_max), branchless single atomics ----------------

template <int D>
__global__ __launch_bounds__(256) void scatter_max_bf16(const bf16* __restrict__ feat,
                                                        const int* __restrict__ src_idx,
                                                        const int* __restrict__ dst_idx,
                                                        int* __restrict__ agg_enc, int nE) {
    constexpr int PER = D / 8;                  // 8 features per thread
    long long gid = (long long)blockIdx.x * blockDim.x + threadIdx.x;
    long long total = (long long)nE * PER;
    if (gid >= total) return;
    int e  = (int)(gid >> (D == 128 ? 4 : 5));  // gid / PER
    int f8 = ((int)gid & (PER - 1)) * 8;
    int s = src_idx[e];
    int d = dst_idx[e];
    bf16x8 v = *(const bf16x8*)(feat + (size_t)s * D + f8);
    int* dp = agg_enc + (size_t)d * D + f8;
    #pragma unroll
    for (int i = 0; i < 8; i++) atomicMax(dp + i, enc_f32((float)v[i]));
}

// ---------------- BatchNorm ----------------

__global__ void bn_zero(float* accS, float* accQ) {
    int c = threadIdx.x;
    accS[c] = 0.0f; accQ[c] = 0.0f;
}

__global__ __launch_bounds__(256) void bn_partial(const float* __restrict__ h,
                                                  float* __restrict__ accS, float* __restrict__ accQ,
                                                  int nrows) {
    int c  = threadIdx.x;                 // column 0..255 (coalesced across block per row)
    int r0 = blockIdx.x * 256;
    int r1 = r0 + 256; if (r1 > nrows) r1 = nrows;
    float s = 0.0f, q = 0.0f;
    for (int r = r0; r < r1; r++) {
        float v = h[(size_t)r * HID + c];
        s += v; q += v * v;
    }
    atomicAdd(&accS[c], s);
    atomicAdd(&accQ[c], q);
}

__global__ void bn_final(const float* accS, const float* accQ,
                         float* meanv, float* rstdv, int nrows) {
    int c = threadIdx.x;
    float inv_n = 1.0f / (float)nrows;
    float m = accS[c] * inv_n;
    float var = accQ[c] * inv_n - m * m;
    meanv[c] = m;
    rstdv[c] = rsqrtf(var + 1e-5f);
}

__global__ __launch_bounds__(256) void bn_apply_bf16(const float* __restrict__ h,
                                                     const float* __restrict__ meanv,
                                                     const float* __restrict__ rstdv,
                                                     const float* __restrict__ gamma,
                                                     const float* __restrict__ beta,
                                                     bf16* __restrict__ out, long long n) {
    long long gid = (long long)blockIdx.x * blockDim.x + threadIdx.x;
    if (gid >= n) return;
    int c = (int)(gid & (HID - 1));
    float v = (h[gid] - meanv[c]) * rstdv[c] * gamma[c] + beta[c];
    out[gid] = (bf16)v;
}

// ---------------- fused SAGE GEMM: out = [A1|A2] @ [B1|B2]^T + bias ----------------
// A1:[N,K1] A2:[N,K2] bf16 row-major; B1:[OUT,K1] B2:[OUT,K2] bf16 row-major.
// Block = 8 waves, 128x128 tile; wave = 64x32 (4x2 WMMA 16x16 tiles), K-chunks of 32.

template <int K1, int K2, int OUT>
__global__ __launch_bounds__(256) void sage_gemm(const bf16* __restrict__ A1, const bf16* __restrict__ A2,
                                                 const bf16* __restrict__ B1, const bf16* __restrict__ B2,
                                                 const float* __restrict__ bias,
                                                 float* __restrict__ outp, int nrows) {
    constexpr int KT = K1 + K2;
    const int lane = threadIdx.x & 31;
    const int wave = threadIdx.x >> 5;
    const int wm = wave & 1;           // 2 waves along M
    const int wn = wave >> 1;          // 4 waves along N
    const int bm = blockIdx.y * 128 + wm * 64;
    const int bn = blockIdx.x * 128 + wn * 32;
    const int l15 = lane & 15;
    const bool hi = lane >= 16;

    v8f c[4][2];
    #pragma unroll
    for (int i = 0; i < 4; i++)
        #pragma unroll
        for (int j = 0; j < 2; j++) c[i][j] = (v8f)0.0f;

    int rows[4];
    #pragma unroll
    for (int i = 0; i < 4; i++) {
        int r = bm + i * 16 + l15;
        rows[i] = r < nrows ? r : (nrows - 1);   // clamp: keep EXEC uniform for WMMA
    }
    int cols[2];
    #pragma unroll
    for (int j = 0; j < 2; j++) cols[j] = bn + j * 16 + l15;  // OUT % 128 == 0, always in range

    for (int kb = 0; kb < KT; kb += 32) {       // uniform source switch at kb == K1
        const bf16* Ab; const bf16* Bb; int kloc, kdim;
        if (kb < K1) { Ab = A1; Bb = B1; kloc = kb;      kdim = K1; }
        else         { Ab = A2; Bb = B2; kloc = kb - K1; kdim = K2; }

        // A fragment: lane holds row rows[i], K = base+{0..7} and base+16+{0..7}, base = hi?8:0
        const int aklo = kloc + (hi ? 8 : 0);
        v16bf a[4];
        #pragma unroll
        for (int i = 0; i < 4; i++) {
            const bf16* p = Ab + (size_t)rows[i] * kdim + aklo;
            bf16x8 lo = *(const bf16x8*)p;
            bf16x8 hh = *(const bf16x8*)(p + 16);
            #pragma unroll
            for (int t = 0; t < 8; t++) { a[i][t] = lo[t]; a[i][8 + t] = hh[t]; }
        }
        // B fragment: lane holds column cols[j], K = base+{0..7} and base+16+{0..7}, base = kloc + (hi?8:0)
        const int bklo = kloc + (hi ? 8 : 0);
        v16bf b[2];
        #pragma unroll
        for (int j = 0; j < 2; j++) {
            const bf16* p = Bb + (size_t)cols[j] * kdim + bklo;
            bf16x8 lo = *(const bf16x8*)p;
            bf16x8 hh = *(const bf16x8*)(p + 16);
            #pragma unroll
            for (int t = 0; t < 8; t++) { b[j][t] = lo[t]; b[j][8 + t] = hh[t]; }
        }
        #pragma unroll
        for (int i = 0; i < 4; i++)
            #pragma unroll
            for (int j = 0; j < 2; j++)
                c[i][j] = __builtin_amdgcn_wmma_f32_16x16x32_bf16(
                    false, a[i], false, b[j], (short)0, c[i][j], false, false);
    }

    // C/D layout: lane column = cols[j]; element v -> row bm + i*16 + (hi?8:0) + v
    float bv[2];
    #pragma unroll
    for (int j = 0; j < 2; j++) bv[j] = bias[cols[j]];

    if (bm + 64 <= nrows) {
        // wave-uniform fast path: whole 64-row wave tile in range, unconditional stores
        #pragma unroll
        for (int i = 0; i < 4; i++) {
            const size_t rbase = (size_t)(bm + i * 16 + (hi ? 8 : 0));
            #pragma unroll
            for (int v = 0; v < 8; v++) {
                float* po = outp + (rbase + v) * OUT;
                #pragma unroll
                for (int j = 0; j < 2; j++) po[cols[j]] = c[i][j][v] + bv[j];
            }
        }
    } else {
        #pragma unroll
        for (int i = 0; i < 4; i++) {
            #pragma unroll
            for (int v = 0; v < 8; v++) {
                int row = bm + i * 16 + (hi ? 8 : 0) + v;
                if (row < nrows) {
                    float* po = outp + (size_t)row * OUT;
                    #pragma unroll
                    for (int j = 0; j < 2; j++) po[cols[j]] = c[i][j][v] + bv[j];
                }
            }
        }
    }
}

// ---------------- driver ----------------

static inline long long cdiv(long long a, long long b) { return (a + b - 1) / b; }

extern "C" void kernel_launch(void* const* d_in, const int* in_sizes, int n_in,
                              void* d_out, int out_size, void* d_ws, size_t ws_size,
                              hipStream_t stream) {
    const float* x    = (const float*)d_in[0];
    const int*   ei   = (const int*)d_in[1];
    const float* W1l  = (const float*)d_in[2];
    const float* b1   = (const float*)d_in[3];
    const float* W1r  = (const float*)d_in[4];
    const float* gamma= (const float*)d_in[5];
    const float* beta = (const float*)d_in[6];
    const float* W2l  = (const float*)d_in[7];
    const float* b2   = (const float*)d_in[8];
    const float* W2r  = (const float*)d_in[9];

    const int N = in_sizes[0] / 128;   // nodes
    const int E = in_sizes[1] / 2;     // edges
    const int* src = ei;
    const int* dst = ei + E;

    // workspace arena (256B aligned slabs, lifetimes overlapped)
    char* ws = (char*)d_ws;
    size_t off = 0;
    auto alloc = [&](size_t bytes) -> char* {
        char* p = ws + off;
        off += (bytes + 255) & ~(size_t)255;
        return p;
    };
    int*   aggi  = (int*)  alloc((size_t)N * 256 * 4); // enc agg1 -> h f32 -> enc agg2
    bf16*  slabB = (bf16*) alloc((size_t)N * 256 * 2); // xb+agg1b -> agg2b
    bf16*  hnb   = (bf16*) alloc((size_t)N * 256 * 2); // normalized h, bf16
    bf16*  w1lb  = (bf16*) alloc(256 * 128 * 2);
    bf16*  w1rb  = (bf16*) alloc(256 * 128 * 2);
    bf16*  w2lb  = (bf16*) alloc(512 * 256 * 2);
    bf16*  w2rb  = (bf16*) alloc(512 * 256 * 2);
    float* accS  = (float*)alloc(256 * 4);
    float* accQ  = (float*)alloc(256 * 4);
    float* meanv = (float*)alloc(256 * 4);
    float* rstdv = (float*)alloc(256 * 4);

    bf16* xb    = slabB;                     // [N,128]
    bf16* agg1b = slabB + (size_t)N * 128;   // [N,128]
    bf16* agg2b = slabB;                     // [N,256] (phase 2 reuse)
    float* h    = (float*)aggi;              // [N,256] (after agg1 dead)

    const long long nx = (long long)N * 128;
    const long long nh = (long long)N * 256;

    // weights + x -> bf16
    cvt_bf16x4<<<cdiv(nx / 4, 256), 256, 0, stream>>>(x, xb, nx / 4);
    cvt_bf16x4<<<cdiv(256 * 128 / 4, 256), 256, 0, stream>>>(W1l, w1lb, 256 * 128 / 4);
    cvt_bf16x4<<<cdiv(256 * 128 / 4, 256), 256, 0, stream>>>(W1r, w1rb, 256 * 128 / 4);
    cvt_bf16x4<<<cdiv(512 * 256 / 4, 256), 256, 0, stream>>>(W2l, w2lb, 512 * 256 / 4);
    cvt_bf16x4<<<cdiv(512 * 256 / 4, 256), 256, 0, stream>>>(W2r, w2rb, 512 * 256 / 4);

    // ---- layer 1: agg = segment_max(x[src], dst) ----
    fill_i32<<<cdiv(nx, 256), 256, 0, stream>>>(aggi, ENC_NEG_INF, nx);
    scatter_max_bf16<128><<<cdiv((long long)E * 16, 256), 256, 0, stream>>>(xb, src, dst, aggi, E);
    cvt_agg_bf16x4<<<cdiv(nx / 4, 256), 256, 0, stream>>>(aggi, agg1b, nx / 4);

    // h = [agg|x] @ [W1l|W1r]^T + b1   (writes over aggi slab; reads only slabB)
    sage_gemm<128, 128, 256><<<dim3(2, (unsigned)cdiv(N, 128)), 256, 0, stream>>>(
        agg1b, xb, w1lb, w1rb, b1, h, N);

    // ---- BatchNorm + cvt to bf16 ----
    bn_zero<<<1, 256, 0, stream>>>(accS, accQ);
    bn_partial<<<cdiv(N, 256), 256, 0, stream>>>(h, accS, accQ, N);
    bn_final<<<1, 256, 0, stream>>>(accS, accQ, meanv, rstdv, N);
    bn_apply_bf16<<<cdiv(nh, 256), 256, 0, stream>>>(h, meanv, rstdv, gamma, beta, hnb, nh);

    // ---- layer 2: agg2 = segment_max(hn[src], dst) ----
    fill_i32<<<cdiv(nh, 256), 256, 0, stream>>>(aggi, ENC_NEG_INF, nh);
    scatter_max_bf16<256><<<cdiv((long long)E * 32, 256), 256, 0, stream>>>(hnb, src, dst, aggi, E);
    cvt_agg_bf16x4<<<cdiv(nh / 4, 256), 256, 0, stream>>>(aggi, agg2b, nh / 4);

    // out = [agg2|hn] @ [W2l|W2r]^T + b2
    sage_gemm<256, 256, 512><<<dim3(4, (unsigned)cdiv(N, 128)), 256, 0, stream>>>(
        agg2b, hnb, w2lb, w2rb, b2, (float*)d_out, N);
}
